// SpectralAttention_13778255085856
// MI455X (gfx1250) — compile-verified, matrix operand order checked by reading, not run on hardware
//
#include <hip/hip_runtime.h>
#include <hip/hip_bf16.h>

typedef _Float16 half_t;
typedef __attribute__((ext_vector_type(16))) _Float16 v16h;
typedef __attribute__((ext_vector_type(8)))  _Float16 v8h;
typedef __attribute__((ext_vector_type(8)))  float    v8f;

#define BATCH   16
#define CDIM    1024
#define NHEAD   8
#define NFEAT   512
#define CH      64
#define BH      (BATCH * NHEAD)      // 128
#define T_TILE  128                  // query rows per block (8 waves x 16)
#define S_TILE  64                   // key columns streamed per chunk
#define NCHUNK  (CDIM / S_TILE)      // 16
#define EPSV    1e-5f

// ---------------------------------------------------------------- kernel 1: BN
__global__ __launch_bounds__(256) void sa_bn_kernel(
    const float* __restrict__ x, const float* __restrict__ g,
    const float* __restrict__ be, const float* __restrict__ mu,
    const float* __restrict__ var, float* __restrict__ xn) {
  int i = blockIdx.x * 256 + threadIdx.x;          // 16384 threads
  int c = i & (CDIM - 1);
  xn[i] = (x[i] - mu[c]) * rsqrtf(var[c] + EPSV) * g[c] + be[c];
}

// ------------------------------------------ kernel 2: 1x3 conv + fp16 repack
// qT,kT: [bh][t][64] (row-major over c) ; v: [bh][c][1024]
__global__ __launch_bounds__(256) void sa_conv_pack_kernel(
    const float* __restrict__ xn, const float* __restrict__ w_in,
    const float* __restrict__ b_in, half_t* __restrict__ qT,
    half_t* __restrict__ kT, half_t* __restrict__ vv) {
  int i = blockIdx.x * 256 + threadIdx.x;          // 16*1536*1024 threads
  int t  = i & 1023;
  int bo = i >> 10;                                // b*1536 + o
  int o  = bo % 1536;
  int b  = bo / 1536;
  const float* xrow = xn + (b << 10);
  float xm = (t > 0)    ? xrow[t - 1] : 0.0f;
  float x0 = xrow[t];
  float xp = (t < 1023) ? xrow[t + 1] : 0.0f;
  float val = fmaf(w_in[o * 3 + 0], xm,
              fmaf(w_in[o * 3 + 1], x0,
              fmaf(w_in[o * 3 + 2], xp, b_in[o])));
  int h = o / 192;
  int r = o - h * 192;
  int j = r >> 6;                                  // 0=q 1=k 2=v
  int c = r & 63;
  int bh = (b << 3) + h;
  half_t hv = (half_t)val;
  if (j == 0)      qT[((bh << 10) + t) * 64 + c] = hv;
  else if (j == 1) kT[((bh << 10) + t) * 64 + c] = hv;
  else             vv[(((bh << 6) + c) << 10) + t] = hv;
}

// -------------------------------------------------------- WMMA fragment tools
__device__ inline v8f wmma32(v16h a, v16h b, v8f c) {
  return __builtin_amdgcn_wmma_f32_16x16x32_f16(
      false, a, false, b, (short)0, c, false, false);
}

// Load one 16x32 f16 A/B fragment from a row-major [row][rowStride] LDS array.
// row must already include (lane&15); hi = lane>>4.
// Halves 0..7  <- K = kb + hi*8 + 0..7
// Halves 8..15 <- K = kb + 16 + hi*8 + 0..7
__device__ inline v16h ld_frag(const half_t* base, int rowStride, int row,
                               int kb, int hiLane) {
  const half_t* p = base + row * rowStride + kb + hiLane * 8;
  v8h a = *(const v8h*)p;
  v8h b = *(const v8h*)(p + 16);
  v16h r;
#pragma unroll
  for (int i = 0; i < 8; ++i) { r[i] = a[i]; r[i + 8] = b[i]; }
  return r;
}

// ---------------------------------------- CDNA5 async memory->LDS (ASYNCcnt)
// LDS aperture is 4GB-aligned: low 32 bits of the generic pointer == LDS offset.
__device__ inline void async_copy_b128(half_t* ldsDst, const half_t* gSrc) {
  unsigned off = (unsigned)(uintptr_t)ldsDst;
  asm volatile("global_load_async_to_lds_b128 %0, %1, off"
               :: "v"(off), "v"(gSrc) : "memory");
}

// Stage one S_TILE chunk of K^T (64x64 halves) and V (64x64 halves).
// Exactly 4 async-load instructions per wave (2 K + 2 V).
__device__ inline void stage_kv_async(half_t* dKT, half_t* dV,
                                      const half_t* kg, const half_t* vg,
                                      int s0, int tid) {
#pragma unroll
  for (int i = 0; i < 2; ++i) {
    int idx = tid + i * 256;                        // 512 x 16B chunks
    async_copy_b128(dKT + idx * 8, kg + s0 * 64 + idx * 8);
  }
#pragma unroll
  for (int i = 0; i < 2; ++i) {
    int idx = tid + i * 256;
    int row = idx >> 3;                             // c row (0..63)
    int chk = idx & 7;                              // 8B-half chunk in row
    async_copy_b128(dV + row * S_TILE + chk * 8,
                    vg + row * 1024 + s0 + chk * 8);
  }
}

// --------------------------------------------- kernel 3: flash attention core
// grid: (1024/T_TILE, BH) = (8, 128); block: 256 threads = 8 waves.
// Double-buffered async K/V staging; each wave owns 16 query rows.
// Epilogue fuses the 512->1 projection: pr[bh][t] = sum_c w_out[h*64+c]*O[t,c].
__global__ __launch_bounds__(256) void sa_flash_attn_kernel(
    const half_t* __restrict__ qT, const half_t* __restrict__ kT,
    const half_t* __restrict__ vv, const float* __restrict__ w_out,
    float* __restrict__ pr) {
  __shared__ __align__(16) half_t sQT[T_TILE * 64];        // 16 KB  Q^T tile
  __shared__ __align__(16) half_t sKT[2][S_TILE * 64];     // 16 KB  K^T dbl-buf
  __shared__ __align__(16) half_t sV [2][64 * S_TILE];     // 16 KB  V   dbl-buf
  __shared__ __align__(16) half_t sP [8 * 16 * S_TILE];    // 16 KB  per-wave P

  const int tb   = blockIdx.x;                 // t block (0..7)
  const int bh   = blockIdx.y;                 // batch*head (0..127)
  const int head = bh & 7;
  const int tid  = threadIdx.x;
  const int wave = tid >> 5;
  const int lane = tid & 31;
  const int n16  = lane & 15;
  const int hiL  = lane >> 4;

  const half_t* kg = kT + (bh << 10) * 64;
  const half_t* vg = vv + (bh << 6) * 1024;

  // prefetch chunk 0 into buffer 0 (async, no VGPR round-trip)
  stage_kv_async(sKT[0], sV[0], kg, vg, 0, tid);

  // ---- stage Q^T tile (contiguous 128 rows x 64 halves = 1024 x uint4)
  {
    const uint4* src = (const uint4*)(qT + ((bh << 10) + tb * T_TILE) * 64);
    uint4* dst = (uint4*)sQT;
#pragma unroll
    for (int i = 0; i < 4; ++i) dst[tid + i * 256] = src[tid + i * 256];
  }
  __syncthreads();

  // Per-wave Q fragments (invariant across S chunks): M=t rows, K=c (2x32).
  v16h qa0 = ld_frag(sQT, 64, wave * 16 + n16, 0,  hiL);
  v16h qa1 = ld_frag(sQT, 64, wave * 16 + n16, 32, hiL);

  v8f O0 = {}, O1 = {}, O2 = {}, O3 = {};      // 16t x 64c accumulators
  float m[8], l[8];
#pragma unroll
  for (int r = 0; r < 8; ++r) { m[r] = -3.0e38f; l[r] = 0.0f; }

  half_t* sPw = sP + wave * 16 * S_TILE;

  for (int it = 0; it < NCHUNK; ++it) {
    const half_t* cKT = sKT[it & 1];
    const half_t* cV  = sV[it & 1];

    if (it + 1 < NCHUNK) {
      // prefetch next chunk into the other buffer (4 async instrs per wave)
      stage_kv_async(sKT[(it + 1) & 1], sV[(it + 1) & 1], kg, vg,
                     (it + 1) * S_TILE, tid);
      // drain chunk-it's 4 loads; leave the new 4 in flight
      asm volatile("s_wait_asynccnt 0x4" ::: "memory");
    } else {
      asm volatile("s_wait_asynccnt 0x0" ::: "memory");
    }
    __syncthreads();                         // chunk-it data visible to all

    // ---- scores: 4 s-tiles, each = 2 WMMAs over c (scale = CH^-0.5 = 1/8)
    v8f S[4];
#pragma unroll
    for (int st = 0; st < 4; ++st) {
      v16h kb0 = ld_frag(cKT, 64, st * 16 + n16, 0,  hiL);
      v16h kb1 = ld_frag(cKT, 64, st * 16 + n16, 32, hiL);
      v8f acc = {};
      acc = wmma32(qa0, kb0, acc);
      acc = wmma32(qa1, kb1, acc);
#pragma unroll
      for (int r = 0; r < 8; ++r) acc[r] *= 0.125f;
      S[st] = acc;
    }

    // ---- online softmax stats (reduce across the 16-lane N group)
    float mnew[8], cf[8], rs[8];
#pragma unroll
    for (int r = 0; r < 8; ++r) {
      float v = S[0][r];
#pragma unroll
      for (int st = 1; st < 4; ++st) v = fmaxf(v, S[st][r]);
#pragma unroll
      for (int msk = 1; msk < 16; msk <<= 1)
        v = fmaxf(v, __shfl_xor(v, msk, 32));
      mnew[r] = fmaxf(m[r], v);
      cf[r] = __expf(m[r] - mnew[r]);
      m[r] = mnew[r];
      rs[r] = 0.0f;
    }
#pragma unroll
    for (int r = 0; r < 8; ++r) {
      O0[r] *= cf[r]; O1[r] *= cf[r]; O2[r] *= cf[r]; O3[r] *= cf[r];
    }

    // ---- P = exp(S - mnew): accumulate row sums, park fp16 P in LDS
#pragma unroll
    for (int st = 0; st < 4; ++st) {
#pragma unroll
      for (int r = 0; r < 8; ++r) {
        float e = __expf(S[st][r] - mnew[r]);
        rs[r] += e;
        sPw[(r + hiL * 8) * S_TILE + st * 16 + n16] = (half_t)e;
      }
    }
#pragma unroll
    for (int r = 0; r < 8; ++r) {
      float v = rs[r];
#pragma unroll
      for (int msk = 1; msk < 16; msk <<= 1) v += __shfl_xor(v, msk, 32);
      l[r] = l[r] * cf[r] + v;
    }

    // ---- O += P x V^T  (K-dim = 64 -> two 32-chunks; 4 c-tiles)
#pragma unroll
    for (int kk = 0; kk < 2; ++kk) {
      v16h pa  = ld_frag(sPw, S_TILE, n16, kk * 32, hiL);
      v16h vb0 = ld_frag(cV,  S_TILE,  0 + n16, kk * 32, hiL);
      v16h vb1 = ld_frag(cV,  S_TILE, 16 + n16, kk * 32, hiL);
      v16h vb2 = ld_frag(cV,  S_TILE, 32 + n16, kk * 32, hiL);
      v16h vb3 = ld_frag(cV,  S_TILE, 48 + n16, kk * 32, hiL);
      O0 = wmma32(pa, vb0, O0);
      O1 = wmma32(pa, vb1, O1);
      O2 = wmma32(pa, vb2, O2);
      O3 = wmma32(pa, vb3, O3);
    }
    __syncthreads();   // all reads of buf[it&1] done before it is re-filled
  }

  // ---- epilogue: normalize + fused 512->1 projection, per-head partial
  float wo0 = w_out[head * 64 +  0 + n16];
  float wo1 = w_out[head * 64 + 16 + n16];
  float wo2 = w_out[head * 64 + 32 + n16];
  float wo3 = w_out[head * 64 + 48 + n16];
  float contrib[8];
#pragma unroll
  for (int r = 0; r < 8; ++r) {
    float v = O0[r] * wo0 + O1[r] * wo1 + O2[r] * wo2 + O3[r] * wo3;
    v *= (1.0f / l[r]);
#pragma unroll
    for (int msk = 1; msk < 16; msk <<= 1) v += __shfl_xor(v, msk, 32);
    contrib[r] = v;
  }
  if (n16 == 0) {
#pragma unroll
    for (int r = 0; r < 8; ++r) {
      int t = tb * T_TILE + wave * 16 + hiL * 8 + r;
      pr[(bh << 10) + t] = contrib[r];
    }
  }
}

// ------------------------------------------------- kernel 4: head-sum + resid
__global__ __launch_bounds__(256) void sa_finalize_kernel(
    const float* __restrict__ xn, const float* __restrict__ pr,
    const float* __restrict__ b_out, float* __restrict__ out) {
  int i = blockIdx.x * 256 + threadIdx.x;        // 16384 threads
  int b = i >> 10;
  int t = i & 1023;
  float s = 0.0f;
#pragma unroll
  for (int h = 0; h < 8; ++h) s += pr[(((b << 3) + h) << 10) + t];
  out[i] = xn[i] + b_out[0] + s;
}

// --------------------------------------------------------------------- launch
extern "C" void kernel_launch(void* const* d_in, const int* in_sizes, int n_in,
                              void* d_out, int out_size, void* d_ws, size_t ws_size,
                              hipStream_t stream) {
  const float* x        = (const float*)d_in[0];
  const float* bn_gamma = (const float*)d_in[1];
  const float* bn_beta  = (const float*)d_in[2];
  const float* bn_mean  = (const float*)d_in[3];
  const float* bn_var   = (const float*)d_in[4];
  const float* w_in     = (const float*)d_in[5];
  const float* b_in     = (const float*)d_in[6];
  const float* w_out    = (const float*)d_in[7];
  const float* b_out    = (const float*)d_in[8];
  float* out = (float*)d_out;

  // workspace layout
  float*  xn = (float*)d_ws;                       // 16384 f32
  float*  pr = xn + BATCH * CDIM;                  // 128*1024 f32
  half_t* qT = (half_t*)(pr + BH * CDIM);          // 128*1024*64 f16
  half_t* kT = qT + (size_t)BH * CDIM * CH;
  half_t* vv = kT + (size_t)BH * CDIM * CH;

  sa_bn_kernel<<<(BATCH * CDIM) / 256, 256, 0, stream>>>(
      x, bn_gamma, bn_beta, bn_mean, bn_var, xn);

  int conv_total = BATCH * 3 * NFEAT * CDIM;       // 25,165,824
  sa_conv_pack_kernel<<<conv_total / 256, 256, 0, stream>>>(
      xn, w_in, b_in, qT, kT, vv);

  dim3 agrid(CDIM / T_TILE, BH);                   // (8, 128)
  sa_flash_attn_kernel<<<agrid, 256, 0, stream>>>(qT, kT, vv, w_out, pr);

  sa_finalize_kernel<<<(BATCH * CDIM) / 256, 256, 0, stream>>>(
      xn, pr, b_out, out);
}
